// SST_21371757265553
// MI455X (gfx1250) — matrix operand order repaired and checked by reading, not compiled, not run on hardware
//
#include <hip/hip_runtime.h>
#include <math.h>

// ---------------- problem constants ----------------
#define N_   98304
#define C_   256
#define H_   8
#define T_   64
#define V_   48
#define W_   2048
#define L_   2
#define F_   1024
#define DH_  32
#define SCALE_ 0.17677669529663687f   // 1/sqrt(32)

// ---------------- vector types ----------------
typedef __attribute__((ext_vector_type(16))) __bf16        v16bf;
typedef __attribute__((ext_vector_type(8)))  float         v8f;
typedef __attribute__((ext_vector_type(8)))  unsigned int  v8u;
typedef __attribute__((ext_vector_type(4)))  __bf16        bf4;

// ---------------- bf16 conversion: native cast (v_cvt on gfx1250) ----------
__device__ __forceinline__ __bf16 f2bf(float f) { return (__bf16)f; }

// ---------------- CDNA5 async memory->LDS copy (ASYNCcnt tracked) ----------
// Low 32 bits of a flat shared-memory address are the LDS byte address
// (ISA 10.2 aperture rules).
__device__ __forceinline__ void async_ld_b64(void* lds_dst, const void* gsrc) {
    unsigned lo = (unsigned)(unsigned long long)lds_dst;
    unsigned long long ga = (unsigned long long)gsrc;
    asm volatile("global_load_async_to_lds_b64 %0, %1, off"
                 :: "v"(lo), "v"(ga) : "memory");
}
// Wait until <= Nw async ops outstanding (async loads complete in order,
// so waiting to the next tile's issue count guarantees the previous tile landed).
template <int Nw>
__device__ __forceinline__ void wait_async_le() {
    asm volatile("s_wait_asynccnt %0" :: "i"(Nw) : "memory");
}

// ---------------- WMMA fragment loaders (16-bit layouts, ISA 7.12.2) -------
__device__ __forceinline__ v16bf frag_a(const __bf16* tile, int m0, int stride, int lane) {
    const unsigned* t32 = (const unsigned*)tile;
    int m  = m0 + (lane & 15);
    int kb = (lane >> 4) << 3;            // 0 or 8
    int base = (m * stride) >> 1;         // stride in bf16 elems (even)
    v8u r;
#pragma unroll
    for (int j = 0; j < 8; ++j) {
        int k0 = (j < 4) ? (kb + 2 * j) : (kb + 16 + 2 * (j - 4));
        r[j] = t32[base + (k0 >> 1)];
    }
    return __builtin_bit_cast(v16bf, r);
}
__device__ __forceinline__ v16bf frag_b(const __bf16* tile, int n0, int stride, int lane) {
    const unsigned* t32 = (const unsigned*)tile;
    int n  = n0 + (lane & 15);
    int kb = (lane >> 4) << 3;            // 0 or 8
    int base = (n * stride) >> 1;
    v8u r;
#pragma unroll
    for (int j = 0; j < 8; ++j) {
        int k0 = (j < 4) ? (kb + 2 * j) : (kb + 16 + 2 * (j - 4));
        r[j] = t32[base + (k0 >> 1)];
    }
    return __builtin_bit_cast(v16bf, r);
}
__device__ __forceinline__ v8f wmma_bf16(v16bf a, v16bf b, v8f c) {
    return __builtin_amdgcn_wmma_f32_16x16x32_bf16(false, a, false, b, (short)0, c, false, false);
}

// ============================================================
// K0: one-time f32 -> bf16 weight conversion
// ============================================================
__global__ void k_cvt(const float* __restrict__ src, __bf16* __restrict__ dst, int n4) {
    int i = blockIdx.x * blockDim.x + threadIdx.x;
    if (i >= n4) return;
    float4 f = ((const float4*)src)[i];
    bf4 b; b[0] = f2bf(f.x); b[1] = f2bf(f.y); b[2] = f2bf(f.z); b[3] = f2bf(f.w);
    ((bf4*)dst)[i] = b;
}

// ============================================================
// K1: gather voxels into windows; qk = x + pos ; cast to bf16
// ============================================================
__global__ void k_gather(const float* __restrict__ feat, const float* __restrict__ pos,
                         const int* __restrict__ idx,
                         __bf16* __restrict__ xw, __bf16* __restrict__ qkw) {
    int w   = blockIdx.x;
    int tid = threadIdx.x;
    const int* widx = idx + (size_t)w * T_;
    for (int i4 = tid; i4 < T_ * C_ / 4; i4 += blockDim.x) {
        int t  = i4 >> 6;
        int c4 = i4 & 63;
        int g  = (t < V_) ? widx[t] : 0;
        float4 f = ((const float4*)(feat + (size_t)g * C_))[c4];
        float4 p = ((const float4*)(pos + ((size_t)w * T_ + t) * C_))[c4];
        bf4 xb, qb;
        xb[0] = f2bf(f.x); xb[1] = f2bf(f.y); xb[2] = f2bf(f.z); xb[3] = f2bf(f.w);
        qb[0] = f2bf(f.x + p.x); qb[1] = f2bf(f.y + p.y);
        qb[2] = f2bf(f.z + p.z); qb[3] = f2bf(f.w + p.w);
        size_t ro = ((size_t)w * T_ + t) * C_;
        ((bf4*)(xw  + ro))[c4] = xb;
        ((bf4*)(qkw + ro))[c4] = qb;
    }
}

// ============================================================
// K2: QKV projection, async double-buffered.
// grid = (W_, 12), block = 128 (4 waves); 8 async b64 per thread per tile
// ============================================================
__global__ void k_qkv(const __bf16* __restrict__ qkw, const __bf16* __restrict__ xw,
                      const __bf16* __restrict__ wqkv, const float* __restrict__ bqkv,
                      __bf16* __restrict__ qb, __bf16* __restrict__ kb, __bf16* __restrict__ vb) {
    __shared__ __bf16 shA[2][T_ * 40];
    __shared__ __bf16 shW[2][64 * 40];
    int w   = blockIdx.x;
    int yt  = blockIdx.y;
    int mat = yt >> 2;                   // 0=q 1=k 2=v
    int nt  = yt & 3;
    const __bf16* A   = (mat < 2 ? qkw : xw) + (size_t)w * T_ * C_;
    const __bf16* Wr  = wqkv + (size_t)yt * 64 * C_;
    __bf16*       out = (mat == 0 ? qb : (mat == 1 ? kb : vb));
    int tid  = threadIdx.x;
    int lane = tid & 31, wv = tid >> 5;

    auto stage = [&](int buf, int k0) {
        for (int i = tid; i < T_ * 8; i += 128) {           // 4 per thread
            int r = i >> 3, cc = (i & 7) << 2;
            async_ld_b64(&shA[buf][r * 40 + cc], A + (size_t)r * C_ + k0 + cc);
        }
        for (int i = tid; i < 64 * 8; i += 128) {           // 4 per thread
            int r = i >> 3, cc = (i & 7) << 2;
            async_ld_b64(&shW[buf][r * 40 + cc], Wr + (size_t)r * C_ + k0 + cc);
        }
    };

    v8f acc[4] = {};
    stage(0, 0);
    for (int ks = 0; ks < 8; ++ks) {
        int buf = ks & 1;
        if (ks < 7) { stage(buf ^ 1, (ks + 1) * 32); wait_async_le<8>(); }
        else        { wait_async_le<0>(); }
        __syncthreads();
        v16bf a = frag_a(shA[buf], wv * 16, 40, lane);
#pragma unroll
        for (int j = 0; j < 4; ++j) {
            v16bf b = frag_b(shW[buf], j * 16, 40, lane);
            acc[j] = wmma_bf16(a, b, acc[j]);
        }
        __syncthreads();
    }
    int nlo = lane & 15, mo = (lane >> 4) << 3;
#pragma unroll
    for (int j = 0; j < 4; ++j) {
        float bvv = bqkv[yt * 64 + j * 16 + nlo];
#pragma unroll
        for (int r = 0; r < 8; ++r) {
            int m = wv * 16 + mo + r;
            out[((size_t)w * T_ + m) * C_ + nt * 64 + j * 16 + nlo] = f2bf(acc[j][r] + bvv);
        }
    }
}

// ============================================================
// K3: windowed attention, one block per (window, head), 128 thr
// ============================================================
__global__ void k_attn(const __bf16* __restrict__ qb, const __bf16* __restrict__ kb,
                       const __bf16* __restrict__ vb, __bf16* __restrict__ ow) {
    __shared__ __bf16 shQ[T_ * 40];
    __shared__ __bf16 shK[T_ * 40];
    __shared__ __bf16 shVT[DH_ * 72];
    __shared__ float  shS[T_ * 68];
    __shared__ __bf16 shP[T_ * 72];
    int wh = blockIdx.x;
    int w = wh >> 3, h = wh & 7;
    int tid = threadIdx.x, lane = tid & 31, wv = tid >> 5;
    const __bf16* qp = qb + (size_t)w * T_ * C_ + h * DH_;
    const __bf16* kp = kb + (size_t)w * T_ * C_ + h * DH_;
    const __bf16* vp = vb + (size_t)w * T_ * C_ + h * DH_;
    for (int i = tid; i < T_ * 8; i += 128) {
        int t = i >> 3, cc = (i & 7) << 2;
        async_ld_b64(shQ + t * 40 + cc, qp + (size_t)t * C_ + cc);
        async_ld_b64(shK + t * 40 + cc, kp + (size_t)t * C_ + cc);
    }
    for (int i = tid; i < T_ * DH_; i += 128) {             // V transposed, scalar
        int t = i >> 5, d = i & 31;
        shVT[d * 72 + t] = vp[(size_t)t * C_ + d];
    }
    wait_async_le<0>();
    __syncthreads();
    v8f sacc[4] = {};
    {
        v16bf a = frag_a(shQ, wv * 16, 40, lane);
#pragma unroll
        for (int j = 0; j < 4; ++j) {
            v16bf b = frag_b(shK, j * 16, 40, lane);
            sacc[j] = wmma_bf16(a, b, sacc[j]);
        }
    }
    int nlo = lane & 15, mo = (lane >> 4) << 3;
#pragma unroll
    for (int j = 0; j < 4; ++j)
#pragma unroll
        for (int r = 0; r < 8; ++r) {
            int s = j * 16 + nlo;
            float val = sacc[j][r] * SCALE_;
            if (s >= V_) val = -1e9f;            // key padding mask
            shS[(wv * 16 + mo + r) * 68 + s] = val;
        }
    __syncthreads();
    if (tid < T_) {                              // row softmax
        float* row = shS + tid * 68;
        float mx = row[0];
        for (int c = 1; c < T_; ++c) mx = fmaxf(mx, row[c]);
        float sum = 0.f;
        for (int c = 0; c < T_; ++c) { float e = __expf(row[c] - mx); row[c] = e; sum += e; }
        float inv = 1.0f / sum;
        for (int c = 0; c < T_; ++c) shP[tid * 72 + c] = f2bf(row[c] * inv);
    }
    __syncthreads();
    v8f oacc[2] = {};
#pragma unroll
    for (int ks = 0; ks < 2; ++ks) {
        v16bf a = frag_a(shP + ks * 32, wv * 16, 72, lane);
#pragma unroll
        for (int j = 0; j < 2; ++j) {
            v16bf b = frag_b(shVT + ks * 32, j * 16, 72, lane);
            oacc[j] = wmma_bf16(a, b, oacc[j]);
        }
    }
#pragma unroll
    for (int j = 0; j < 2; ++j)
#pragma unroll
        for (int r = 0; r < 8; ++r) {
            int m = wv * 16 + mo + r;
            ow[((size_t)w * T_ + m) * C_ + h * DH_ + j * 16 + nlo] = f2bf(oacc[j][r]);
        }
}

// ============================================================
// K4: out-proj GEMM + bias + residual scatter, async double-buffered.
// grid = W_, block = 256 (8 waves); 10 async b64 per thread per tile
// ============================================================
__global__ void k_outproj(const __bf16* __restrict__ ow, const __bf16* __restrict__ opw,
                          const float* __restrict__ opb, const int* __restrict__ idx,
                          const float* __restrict__ fin, float* __restrict__ fm) {
    __shared__ __bf16 shA[2][T_ * 40];
    __shared__ __bf16 shW[2][C_ * 40];
    int w = blockIdx.x;
    int tid = threadIdx.x, lane = tid & 31, wv = tid >> 5;
    const __bf16* A = ow + (size_t)w * T_ * C_;
    const int* widx = idx + (size_t)w * T_;
    int mrow0 = (wv & 3) * 16;
    int nbase = (wv >> 2) * 128;

    auto stage = [&](int buf, int k0) {
        for (int i = tid; i < T_ * 8; i += 256) {           // 2 per thread
            int r = i >> 3, cc = (i & 7) << 2;
            async_ld_b64(&shA[buf][r * 40 + cc], A + (size_t)r * C_ + k0 + cc);
        }
        for (int i = tid; i < C_ * 8; i += 256) {           // 8 per thread
            int r = i >> 3, cc = (i & 7) << 2;
            async_ld_b64(&shW[buf][r * 40 + cc], opw + (size_t)r * C_ + k0 + cc);
        }
    };

    v8f acc[8] = {};
    stage(0, 0);
    for (int ks = 0; ks < 8; ++ks) {
        int buf = ks & 1;
        if (ks < 7) { stage(buf ^ 1, (ks + 1) * 32); wait_async_le<10>(); }
        else        { wait_async_le<0>(); }
        __syncthreads();
        v16bf a = frag_a(shA[buf], mrow0, 40, lane);
#pragma unroll
        for (int j = 0; j < 8; ++j) {
            v16bf b = frag_b(shW[buf], nbase + j * 16, 40, lane);
            acc[j] = wmma_bf16(a, b, acc[j]);
        }
        __syncthreads();
    }
    int nlo = lane & 15, mo = (lane >> 4) << 3;
#pragma unroll
    for (int j = 0; j < 8; ++j) {
        int n = nbase + j * 16 + nlo;
        float bvv = opb[n];
#pragma unroll
        for (int r = 0; r < 8; ++r) {
            int t = mrow0 + mo + r;
            if (t < V_) {
                int g = widx[t];
                fm[(size_t)g * C_ + n] = acc[j][r] + bvv + fin[(size_t)g * C_ + n];
            }
        }
    }
}

// ============================================================
// LayerNorm over C_, in place (float4 vectorized); optional bf16 copy
// ============================================================
__global__ void k_layernorm(float* __restrict__ x, const float* __restrict__ s,
                            const float* __restrict__ b, __bf16* __restrict__ xbf) {
    int r = blockIdx.x * blockDim.x + threadIdx.x;
    if (r >= N_) return;
    float4* row = (float4*)(x + (size_t)r * C_);
    float sum = 0.f, sq = 0.f;
    for (int c = 0; c < C_ / 4; ++c) {
        float4 t = row[c];
        sum += t.x + t.y + t.z + t.w;
        sq  += t.x * t.x + t.y * t.y + t.z * t.z + t.w * t.w;
    }
    float mean = sum * (1.0f / C_);
    float var  = sq * (1.0f / C_) - mean * mean;
    float inv  = rsqrtf(var + 1e-5f);
    for (int c = 0; c < C_ / 4; ++c) {
        float4 t = row[c];
        float4 sv = ((const float4*)s)[c];
        float4 bv = ((const float4*)b)[c];
        float4 o;
        o.x = (t.x - mean) * inv * sv.x + bv.x;
        o.y = (t.y - mean) * inv * sv.y + bv.y;
        o.z = (t.z - mean) * inv * sv.z + bv.z;
        o.w = (t.w - mean) * inv * sv.w + bv.w;
        row[c] = o;
        if (xbf) {
            bf4 ob; ob[0] = f2bf(o.x); ob[1] = f2bf(o.y); ob[2] = f2bf(o.z); ob[3] = f2bf(o.w);
            ((bf4*)(xbf + (size_t)r * C_))[c] = ob;
        }
    }
}

// ============================================================
// K5: FFN lin1 + GELU, async double-buffered.
// grid = (N_/64, F_/64), block = 128; 8 async b64 per thread per tile
// ============================================================
__global__ void k_ffn1(const __bf16* __restrict__ fmb, const __bf16* __restrict__ l1w,
                       const float* __restrict__ l1b, __bf16* __restrict__ hbuf) {
    __shared__ __bf16 shA[2][64 * 40];
    __shared__ __bf16 shW[2][64 * 40];
    int m0 = blockIdx.x * 64;
    int n0 = blockIdx.y * 64;
    int tid = threadIdx.x, lane = tid & 31, wv = tid >> 5;
    const __bf16* A  = fmb + (size_t)m0 * C_;
    const __bf16* Wr = l1w + (size_t)n0 * C_;

    auto stage = [&](int buf, int k0) {
        for (int i = tid; i < 64 * 8; i += 128) {
            int r = i >> 3, cc = (i & 7) << 2;
            async_ld_b64(&shA[buf][r * 40 + cc], A + (size_t)r * C_ + k0 + cc);
        }
        for (int i = tid; i < 64 * 8; i += 128) {
            int r = i >> 3, cc = (i & 7) << 2;
            async_ld_b64(&shW[buf][r * 40 + cc], Wr + (size_t)r * C_ + k0 + cc);
        }
    };

    v8f acc[4] = {};
    stage(0, 0);
    for (int ks = 0; ks < 8; ++ks) {
        int buf = ks & 1;
        if (ks < 7) { stage(buf ^ 1, (ks + 1) * 32); wait_async_le<8>(); }
        else        { wait_async_le<0>(); }
        __syncthreads();
        v16bf a = frag_a(shA[buf], wv * 16, 40, lane);
#pragma unroll
        for (int j = 0; j < 4; ++j) {
            v16bf b = frag_b(shW[buf], j * 16, 40, lane);
            acc[j] = wmma_bf16(a, b, acc[j]);
        }
        __syncthreads();
    }
    int nlo = lane & 15, mo = (lane >> 4) << 3;
#pragma unroll
    for (int j = 0; j < 4; ++j) {
        float bvv = l1b[n0 + j * 16 + nlo];
#pragma unroll
        for (int r = 0; r < 8; ++r) {
            int m = wv * 16 + mo + r;
            float xv = acc[j][r] + bvv;
            float gl = 0.5f * xv * (1.0f + erff(xv * 0.70710678118f));   // exact GELU
            hbuf[(size_t)(m0 + m) * F_ + n0 + j * 16 + nlo] = f2bf(gl);
        }
    }
}

// ============================================================
// K6: FFN lin2 + bias + residual (pre-LN), async double-buffered.
// grid = N_/64, block = 256; 10 async b64 per thread per tile
// ============================================================
__global__ void k_ffn2(const __bf16* __restrict__ hbuf, const __bf16* __restrict__ l2w,
                       const float* __restrict__ l2b, const float* __restrict__ fm,
                       float* __restrict__ fout) {
    __shared__ __bf16 shA[2][64 * 40];
    __shared__ __bf16 shW[2][C_ * 40];
    int m0 = blockIdx.x * 64;
    int tid = threadIdx.x, lane = tid & 31, wv = tid >> 5;
    const __bf16* A = hbuf + (size_t)m0 * F_;
    int mrow0 = (wv & 3) * 16;
    int nbase = (wv >> 2) * 128;

    auto stage = [&](int buf, int k0) {
        for (int i = tid; i < 64 * 8; i += 256) {
            int r = i >> 3, cc = (i & 7) << 2;
            async_ld_b64(&shA[buf][r * 40 + cc], A + (size_t)r * F_ + k0 + cc);
        }
        for (int i = tid; i < C_ * 8; i += 256) {
            int r = i >> 3, cc = (i & 7) << 2;
            async_ld_b64(&shW[buf][r * 40 + cc], l2w + (size_t)r * F_ + k0 + cc);
        }
    };

    v8f acc[8] = {};
    stage(0, 0);
    for (int ks = 0; ks < 32; ++ks) {
        int buf = ks & 1;
        if (ks < 31) { stage(buf ^ 1, (ks + 1) * 32); wait_async_le<10>(); }
        else         { wait_async_le<0>(); }
        __syncthreads();
        v16bf a = frag_a(shA[buf], mrow0, 40, lane);
#pragma unroll
        for (int j = 0; j < 8; ++j) {
            v16bf b = frag_b(shW[buf], nbase + j * 16, 40, lane);
            acc[j] = wmma_bf16(a, b, acc[j]);
        }
        __syncthreads();
    }
    int nlo = lane & 15, mo = (lane >> 4) << 3;
#pragma unroll
    for (int j = 0; j < 8; ++j) {
        int n = nbase + j * 16 + nlo;
        float bvv = l2b[n];
#pragma unroll
        for (int r = 0; r < 8; ++r) {
            int row = m0 + mrow0 + mo + r;
            fout[(size_t)row * C_ + n] = acc[j][r] + bvv + fm[(size_t)row * C_ + n];
        }
    }
}

// ============================================================
// host launcher
// ============================================================
extern "C" void kernel_launch(void* const* d_in, const int* in_sizes, int n_in,
                              void* d_out, int out_size, void* d_ws, size_t ws_size,
                              hipStream_t stream) {
    const float* feat  = (const float*)d_in[0];
    const float* pos   = (const float*)d_in[1];
    const int*   winds = (const int*)d_in[2];
    // d_in[3] = padding_mask (implied by t >= V_; unused)
    const float* ipw = (const float*)d_in[4];
    const float* ipb = (const float*)d_in[5];
    const float* opw = (const float*)d_in[6];
    const float* opb = (const float*)d_in[7];
    const float* l1w = (const float*)d_in[8];
    const float* l1b = (const float*)d_in[9];
    const float* l2w = (const float*)d_in[10];
    const float* l2b = (const float*)d_in[11];
    const float* n1s = (const float*)d_in[12];
    const float* n1b = (const float*)d_in[13];
    const float* n2s = (const float*)d_in[14];
    const float* n2b = (const float*)d_in[15];

    char* ws = (char*)d_ws;
    const size_t WTC = (size_t)W_ * T_ * C_;
    __bf16* qkw = (__bf16*)(ws);                     // dead after k_qkv
    __bf16* xw  = (__bf16*)(ws + WTC * 2);           // dead after k_qkv
    __bf16* qb  = (__bf16*)(ws + WTC * 4);
    __bf16* kb  = (__bf16*)(ws + WTC * 6);
    __bf16* vb  = (__bf16*)(ws + WTC * 8);
    __bf16* ow  = qkw;                               // reuse
    float*  fm  = (float*)(ws + WTC * 10);           // [N,C] f32
    __bf16* fmb = xw;                                // reuse, [N,C] bf16
    __bf16* hb  = (__bf16*)(ws + WTC * 10 + (size_t)N_ * C_ * 4);   // [N,F] bf16
    float*  fl  = (float*)(ws + WTC * 10 + (size_t)N_ * C_ * 4 + (size_t)N_ * F_ * 2);
    // bf16 weight caches (converted once per call)
    __bf16* ipw_b = (__bf16*)(ws + WTC * 10 + (size_t)N_ * C_ * 4 + (size_t)N_ * F_ * 2
                                 + (size_t)N_ * C_ * 4);
    __bf16* opw_b = ipw_b + (size_t)L_ * 3 * C_ * C_;
    __bf16* l1w_b = opw_b + (size_t)L_ * C_ * C_;
    __bf16* l2w_b = l1w_b + (size_t)L_ * F_ * C_;

    // convert all weights (both layers) to bf16 once
    {
        int n4;
        n4 = L_ * 3 * C_ * C_ / 4;
        k_cvt<<<(n4 + 255) / 256, 256, 0, stream>>>(ipw, ipw_b, n4);
        n4 = L_ * C_ * C_ / 4;
        k_cvt<<<(n4 + 255) / 256, 256, 0, stream>>>(opw, opw_b, n4);
        n4 = L_ * F_ * C_ / 4;
        k_cvt<<<(n4 + 255) / 256, 256, 0, stream>>>(l1w, l1w_b, n4);
        n4 = L_ * C_ * F_ / 4;
        k_cvt<<<(n4 + 255) / 256, 256, 0, stream>>>(l2w, l2w_b, n4);
    }

    for (int l = 0; l < L_; ++l) {
        const float* fin  = (l == 0) ? feat : fl;
        float*       fout = (l == L_ - 1) ? (float*)d_out : fl;
        const float* posl = pos + (size_t)l * W_ * T_ * C_;
        const int*   il   = winds + (size_t)l * W_ * T_;

        k_gather <<<W_, 256, 0, stream>>>(fin, posl, il, xw, qkw);
        k_qkv    <<<dim3(W_, 12), 128, 0, stream>>>(qkw, xw,
                    ipw_b + (size_t)l * 3 * C_ * C_, ipb + (size_t)l * 3 * C_, qb, kb, vb);
        k_attn   <<<W_ * H_, 128, 0, stream>>>(qb, kb, vb, ow);
        k_outproj<<<W_, 256, 0, stream>>>(ow, opw_b + (size_t)l * C_ * C_,
                    opb + (size_t)l * C_, il, fin, fm);
        k_layernorm<<<(N_ + 255) / 256, 256, 0, stream>>>(fm, n1s + l * C_, n1b + l * C_, fmb);
        k_ffn1   <<<dim3(N_ / 64, F_ / 64), 128, 0, stream>>>(fmb,
                    l1w_b + (size_t)l * F_ * C_, l1b + (size_t)l * F_, hb);
        k_ffn2   <<<N_ / 64, 256, 0, stream>>>(hb, l2w_b + (size_t)l * C_ * F_,
                    l2b + (size_t)l * C_, fm, fout);
        k_layernorm<<<(N_ + 255) / 256, 256, 0, stream>>>(fout, n2s + l * C_, n2b + l * C_, nullptr);
    }
}
